// TreeVAE_15247133901646
// MI455X (gfx1250) — compile-verified, matrix operand order checked by reading, not run on hardware
//
#include <hip/hip_runtime.h>
#include <math.h>

// ---------------------------------------------------------------------------
// Types for CDNA5 WMMA (gfx1250, wave32)
// ---------------------------------------------------------------------------
typedef __attribute__((ext_vector_type(16))) __bf16   v16bf;
typedef __attribute__((ext_vector_type(8)))  float    v8f;
typedef __attribute__((ext_vector_type(8)))  unsigned v8u;

#define EPSV 1e-7f

// ---------------------------------------------------------------------------
// bf16 <-> f32 helpers (storage as unsigned short, RNE rounding)
// ---------------------------------------------------------------------------
__device__ __forceinline__ unsigned short f2bf(float f) {
    unsigned u = __builtin_bit_cast(unsigned, f);
    unsigned r = u + 0x7fffu + ((u >> 16) & 1u);
    return (unsigned short)(r >> 16);
}
__device__ __forceinline__ float bf2f(unsigned short h) {
    unsigned u = ((unsigned)h) << 16;
    return __builtin_bit_cast(float, u);
}

// ---------------------------------------------------------------------------
// Deterministic counter-based Gaussian (hash + Box-Muller)
// ---------------------------------------------------------------------------
__device__ __forceinline__ unsigned ihash(unsigned x) {
    x ^= x >> 16; x *= 0x7feb352dU;
    x ^= x >> 15; x *= 0x846ca68bU;
    x ^= x >> 16; return x;
}
__device__ __forceinline__ float gauss_rng(unsigned k, unsigned i, unsigned j) {
    unsigned s = ihash((k * 0x9E3779B9u) ^ (i * 0x85EBCA6Bu) ^ (j * 0xC2B2AE35u) ^ 0x27D4EB2Fu);
    unsigned a = ihash(s ^ 0x68BC21EBu);
    unsigned b = ihash(s + 0x02E5BE93u);
    float u1 = ((float)a + 1.0f) * (1.0f / 4294967296.0f);
    float u2 = ((float)b) * (1.0f / 4294967296.0f);
    return sqrtf(-2.0f * logf(u1)) * cosf(6.28318530718f * u2);
}

// ---------------------------------------------------------------------------
// Conversion kernels
// ---------------------------------------------------------------------------
// f32 [rows x cols] -> bf16 [rowsPad x colsPad], zero pad (row-major, no transpose)
__global__ void cvt_pad_k(const float* __restrict__ src, unsigned short* __restrict__ dst,
                          int rows, int cols, int rowsPad, int colsPad) {
    long i = (long)blockIdx.x * blockDim.x + threadIdx.x;
    long total = (long)rowsPad * colsPad;
    if (i >= total) return;
    int r = (int)(i / colsPad), c = (int)(i % colsPad);
    float v = (r < rows && c < cols) ? src[(long)r * cols + c] : 0.0f;
    dst[i] = f2bf(v);
}
// batched f32 W[b][K x N] -> bf16 Wt[b][Npad x Kpad] (transposed + zero pad)
__global__ void cvt_trans_k(const float* __restrict__ src, unsigned short* __restrict__ dst,
                            int B, int K, int N, int Kp, int Np) {
    long i = (long)blockIdx.x * blockDim.x + threadIdx.x;
    long per = (long)Np * Kp;
    long total = (long)B * per;
    if (i >= total) return;
    int b = (int)(i / per);
    long r = i - (long)b * per;
    int n = (int)(r / Kp), k = (int)(r % Kp);
    float v = (n < N && k < K) ? src[(long)b * K * N + (long)k * N + n] : 0.0f;
    dst[i] = f2bf(v);
}
// batched f32 bias[b][N] -> f32 [b][Np] zero padded
__global__ void bias_pad_k(const float* __restrict__ src, float* __restrict__ dst,
                           int B, int N, int Np) {
    long i = (long)blockIdx.x * blockDim.x + threadIdx.x;
    long total = (long)B * Np;
    if (i >= total) return;
    int b = (int)(i / Np), n = (int)(i % Np);
    dst[i] = (n < N) ? src[(long)b * N + n] : 0.0f;
}

// ---------------------------------------------------------------------------
// WMMA bf16 GEMM:  C[M x N] = epi(A[M x K] @ Wt^T + bias),  Wt is [N x K] bf16
// Block tile 128x64, 8 waves (4x2) of 32x32, BK=32.
// EPI: 0 = none -> f32, 1 = leaky_relu(0.3) -> bf16,
//      2 = softplus -> f32, 3 = sigmoid -> f32
// Requires: M % 128 == 0, N % 64 == 0, K % 32 == 0, 16B-aligned pointers.
// ---------------------------------------------------------------------------
template <int EPI>
__global__ __launch_bounds__(256) void gemm_bf16_k(
    const unsigned short* __restrict__ A,
    const unsigned short* __restrict__ Wt,
    const float* __restrict__ bias,
    void* __restrict__ Cout,
    int M, int N, int K)
{
    constexpr int BM = 128, BN = 64, BK = 32, LDA = 40, LDB = 40;
    __shared__ unsigned short Alds[BM * LDA];
    __shared__ unsigned short Blds[BN * LDB];   // [n][k]

    const int tid  = threadIdx.x;
    const int lane = tid & 31;
    const int wv   = tid >> 5;
    const int wm   = wv & 3;      // wave row (4 waves x 32 rows)
    const int wn   = wv >> 2;     // wave col (2 waves x 32 cols)
    const long bm  = (long)blockIdx.x * BM;
    const int  bn  = blockIdx.y * BN;

    v8f acc[2][2];
    #pragma unroll
    for (int i = 0; i < 2; i++)
        #pragma unroll
        for (int j = 0; j < 2; j++)
            #pragma unroll
            for (int r = 0; r < 8; r++) acc[i][j][r] = 0.0f;

    const int sr = tid >> 2;            // 0..63
    const int sc = (tid & 3) << 3;      // 0,8,16,24
    const int mrow = wm * 32 + (lane & 15);
    const int ncol = wn * 32 + (lane & 15);
    const int kb   = (lane < 16) ? 0 : 8;   // 16-bit operand lane layout

    const unsigned short* aP0 = &A[(bm + sr) * (long)K + sc];
    const unsigned short* aP1 = &A[(bm + sr + 64) * (long)K + sc];
    const unsigned short* bP  = &Wt[(size_t)(bn + sr) * K + sc];

    for (int k0 = 0; k0 < K; k0 += BK) {
        __syncthreads();
        // ---- stage A tile (128 x 32) and B tile (64 x 32): all 16B vector ops ----
        *(uint4*)(&Alds[sr * LDA + sc])        = *(const uint4*)(aP0 + k0);
        *(uint4*)(&Alds[(sr + 64) * LDA + sc]) = *(const uint4*)(aP1 + k0);
        *(uint4*)(&Blds[sr * LDB + sc])        = *(const uint4*)(bP + k0);
        if (k0 + BK < K) {   // prefetch next tiles (global_prefetch_b8)
            __builtin_prefetch(aP0 + k0 + BK, 0, 0);
            __builtin_prefetch(aP1 + k0 + BK, 0, 0);
            __builtin_prefetch(bP + k0 + BK, 0, 0);
        }
        __syncthreads();

        // ---- fragments (elems 0..7: K=kb..kb+7, 8..15: K=kb+16..kb+23) ----
        v16bf afr[2], bfr[2];
        #pragma unroll
        for (int mi = 0; mi < 2; mi++) {
            const unsigned short* p = &Alds[(mrow + mi * 16) * LDA + kb];
            uint4 lo = *(const uint4*)p;
            uint4 hi = *(const uint4*)(p + 16);
            v8u u; u[0]=lo.x; u[1]=lo.y; u[2]=lo.z; u[3]=lo.w;
                   u[4]=hi.x; u[5]=hi.y; u[6]=hi.z; u[7]=hi.w;
            afr[mi] = __builtin_bit_cast(v16bf, u);
        }
        #pragma unroll
        for (int ni = 0; ni < 2; ni++) {
            const unsigned short* p = &Blds[(ncol + ni * 16) * LDB + kb];
            uint4 lo = *(const uint4*)p;
            uint4 hi = *(const uint4*)(p + 16);
            v8u u; u[0]=lo.x; u[1]=lo.y; u[2]=lo.z; u[3]=lo.w;
                   u[4]=hi.x; u[5]=hi.y; u[6]=hi.z; u[7]=hi.w;
            bfr[ni] = __builtin_bit_cast(v16bf, u);
        }
        #pragma unroll
        for (int mi = 0; mi < 2; mi++)
            #pragma unroll
            for (int ni = 0; ni < 2; ni++)
                acc[mi][ni] = __builtin_amdgcn_wmma_f32_16x16x32_bf16(
                    false, afr[mi], false, bfr[ni],
                    (short)0, acc[mi][ni], false, false);
    }

    // ---- epilogue (unguarded: N is padded to a tile multiple) ----
    const int rbase = (lane < 16) ? 0 : 8;
    const int nloc  = lane & 15;
    #pragma unroll
    for (int mi = 0; mi < 2; mi++) {
        #pragma unroll
        for (int ni = 0; ni < 2; ni++) {
            int col = bn + wn * 32 + ni * 16 + nloc;
            float bv = bias ? bias[col] : 0.0f;
            #pragma unroll
            for (int r = 0; r < 8; r++) {
                long row = bm + wm * 32 + mi * 16 + rbase + r;
                float v = acc[mi][ni][r] + bv;
                if (EPI == 1) {
                    v = v > 0.0f ? v : 0.3f * v;
                    ((unsigned short*)Cout)[row * N + col] = f2bf(v);
                } else {
                    if (EPI == 2) v = fmaxf(v, 0.0f) + log1pf(expf(-fabsf(v)));
                    else if (EPI == 3) v = 1.0f / (1.0f + expf(-v));
                    ((float*)Cout)[row * N + col] = v;
                }
            }
        }
    }
}

// ---------------------------------------------------------------------------
// Warp-per-row dot: out[i] = sigmoid( sum_j bf2f(act[i,j]) * w[j] + b[0] )
// ---------------------------------------------------------------------------
__global__ void rowdot_sigmoid_k(const unsigned short* __restrict__ act,
                                 const float* __restrict__ w,
                                 const float* __restrict__ b,
                                 float* __restrict__ out, int M, int K)
{
    int lane = threadIdx.x & 31;
    int row  = blockIdx.x * (blockDim.x >> 5) + (threadIdx.x >> 5);
    if (row >= M) return;
    const unsigned short* a = act + (long)row * K;
    float s = 0.0f;
    for (int j = lane; j < K; j += 32) s += bf2f(a[j]) * w[j];
    #pragma unroll
    for (int off = 16; off > 0; off >>= 1) s += __shfl_xor(s, off, 32);
    if (lane == 0) out[row] = 1.0f / (1.0f + expf(-(s + b[0])));
}

// ---------------------------------------------------------------------------
// Per-node elementwise: posterior fusion, KL row-sum + clip, reparam sample
// mu/sig buffers have leading dimension 64 (first 32 columns valid).
// ---------------------------------------------------------------------------
__global__ void node_stats_k(const float* __restrict__ muq_hat,
                             const float* __restrict__ sq_hat,
                             const float* __restrict__ mup,     // null for root
                             const float* __restrict__ spv,     // null for root
                             const float* __restrict__ prob,    // null => 1
                             float* __restrict__ kl_out,        // root: =, else: +=
                             unsigned short* __restrict__ z_out,
                             int k, int root, int M)
{
    int i = blockIdx.x * blockDim.x + threadIdx.x;
    if (i >= M) return;
    float kl = 0.0f;
    const long base = (long)i * 64;
    const long zbase = (long)i * 32;
    for (int j = 0; j < 32; j++) {
        float mqh = muq_hat[base + j];
        float sqh = sq_hat[base + j];
        float mp, sp, sq, mq;
        if (mup) {
            mp = mup[base + j];
            sp = spv[base + j];
            sq = 1.0f / (1.0f / sqh + 1.0f / sp);
            mq = sq * (mqh / sqh + mp / sp);
        } else {
            mp = 0.0f; sp = 1.0f; sq = sqh; mq = mqh;
        }
        float vq = sq + EPSV, vp = sp + EPSV;
        float dm = mq - mp;
        kl += 0.5f * (vq / vp + dm * dm / vp - 1.0f + logf(vp / vq));
        float z = mq + sqrtf(vq) * gauss_rng((unsigned)k, (unsigned)i, (unsigned)j);
        z_out[zbase + j] = f2bf(z);
    }
    float p   = prob ? prob[i] : 1.0f;
    float kln = fminf(fmaxf(p * kl, -1.0f), 1000.0f);
    if (root) kl_out[i] = kln;
    else      kl_out[i] += kln;
}

// ---------------------------------------------------------------------------
// Router combine: decision KL + child probabilities
// ---------------------------------------------------------------------------
__global__ void router_k(const float* __restrict__ pl, const float* __restrict__ plq,
                         const float* __restrict__ prob, float* __restrict__ kldec,
                         float* __restrict__ pleft, float* __restrict__ pright, int M)
{
    int i = blockIdx.x * blockDim.x + threadIdx.x;
    if (i >= M) return;
    float p = prob ? prob[i] : 1.0f;
    float a = plq[i], b = pl[i];
    float kd = a * logf(EPSV + a / (b + EPSV)) +
               (1.0f - a) * logf(EPSV + (1.0f - a) / (1.0f - b + EPSV));
    kldec[i] += p * kd;
    pleft[i]  = p * a;
    pright[i] = p * (1.0f - a);
}

// ---------------------------------------------------------------------------
// Warp-per-row BCE accumulate: rec[i] += prob[i] * bce(x[i,:], r[i,:])
// r has leading dimension ldr (832), first N (784) columns valid.
// ---------------------------------------------------------------------------
__global__ void bce_k(const float* __restrict__ x, const float* __restrict__ r,
                      const float* __restrict__ prob, float* __restrict__ rec,
                      int M, int N, int ldr)
{
    int lane = threadIdx.x & 31;
    int row  = blockIdx.x * (blockDim.x >> 5) + (threadIdx.x >> 5);
    if (row >= M) return;
    const float* xr = x + (long)row * N;
    const float* rr = r + (long)row * ldr;
    float s = 0.0f;
    for (int j = lane; j < N; j += 32) {
        float xv = xr[j], rv = rr[j];
        s += xv * logf(rv + EPSV) + (1.0f - xv) * logf(1.0f - rv + EPSV);
    }
    #pragma unroll
    for (int off = 16; off > 0; off >>= 1) s += __shfl_xor(s, off, 32);
    if (lane == 0) rec[row] += prob[row] * (-s);
}

// ---------------------------------------------------------------------------
// Small utility kernels
// ---------------------------------------------------------------------------
__global__ void zero3_k(float* a, float* b, float* c, int n) {
    int i = blockIdx.x * blockDim.x + threadIdx.x;
    if (i < n) { a[i] = 0.0f; b[i] = 0.0f; c[i] = 0.0f; }
}
__global__ void final_k(const float* a, const float* b, const float* c,
                        const float* d, float* out, int n) {
    int i = blockIdx.x * blockDim.x + threadIdx.x;
    if (i < n) out[i] = a[i] + b[i] + c[i] + d[i];
}

// ---------------------------------------------------------------------------
// Host side
// ---------------------------------------------------------------------------
static void gemm(int epi, const unsigned short* A, const unsigned short* Wt,
                 const float* bias, void* C, int M, int N, int K, hipStream_t s)
{
    dim3 g(M / 128, N / 64), b(256);
    switch (epi) {
        case 0: gemm_bf16_k<0><<<g, b, 0, s>>>(A, Wt, bias, C, M, N, K); break;
        case 1: gemm_bf16_k<1><<<g, b, 0, s>>>(A, Wt, bias, C, M, N, K); break;
        case 2: gemm_bf16_k<2><<<g, b, 0, s>>>(A, Wt, bias, C, M, N, K); break;
        default: gemm_bf16_k<3><<<g, b, 0, s>>>(A, Wt, bias, C, M, N, K); break;
    }
}
static void cvt_trans(const float* s, unsigned short* d, int B, int K, int N,
                      int Kp, int Np, hipStream_t st) {
    long n = (long)B * Np * Kp;
    cvt_trans_k<<<(unsigned)((n + 255) / 256), 256, 0, st>>>(s, d, B, K, N, Kp, Np);
}
static void bias_pad(const float* s, float* d, int B, int N, int Np, hipStream_t st) {
    long n = (long)B * Np;
    bias_pad_k<<<(unsigned)((n + 255) / 256), 256, 0, st>>>(s, d, B, N, Np);
}

extern "C" void kernel_launch(void* const* d_in, const int* in_sizes, int n_in,
                              void* d_out, int out_size, void* d_ws, size_t ws_size,
                              hipStream_t stream)
{
    (void)in_sizes; (void)n_in; (void)out_size; (void)ws_size;
    constexpr int M = 16384, DIN = 784, DINP = 800, DOUTP = 832, H = 512, L = 32, LP = 64;

    // ---- inputs (setup_inputs order) ----
    const float* x         = (const float*)d_in[0];
    const float* enc_W0    = (const float*)d_in[1];
    const float* enc_b0    = (const float*)d_in[2];
    const float* enc_W     = (const float*)d_in[3];
    const float* enc_b     = (const float*)d_in[4];
    const float* dense_Wmu = (const float*)d_in[5];
    const float* dense_bmu = (const float*)d_in[6];
    const float* dense_Wsg = (const float*)d_in[7];
    const float* dense_bsg = (const float*)d_in[8];
    const float* trans_Wh  = (const float*)d_in[9];
    const float* trans_bh  = (const float*)d_in[10];
    const float* trans_Wmu = (const float*)d_in[11];
    const float* trans_bmu = (const float*)d_in[12];
    const float* trans_Wsg = (const float*)d_in[13];
    const float* trans_bsg = (const float*)d_in[14];
    const float* routp_W1  = (const float*)d_in[15];
    const float* routp_b1  = (const float*)d_in[16];
    const float* routp_W2  = (const float*)d_in[17];
    const float* routp_b2  = (const float*)d_in[18];
    const float* routp_W3  = (const float*)d_in[19];
    const float* routp_b3  = (const float*)d_in[20];
    const float* routq_W1  = (const float*)d_in[21];
    const float* routq_b1  = (const float*)d_in[22];
    const float* routq_W2  = (const float*)d_in[23];
    const float* routq_b2  = (const float*)d_in[24];
    const float* routq_W3  = (const float*)d_in[25];
    const float* routq_b3  = (const float*)d_in[26];
    const float* dec_W1    = (const float*)d_in[27];
    const float* dec_b1    = (const float*)d_in[28];
    const float* dec_W2    = (const float*)d_in[29];
    const float* dec_b2    = (const float*)d_in[30];

    // ---- workspace arena ----
    char* ws = (char*)d_ws;
    size_t off = 0;
    auto alloc = [&](size_t bytes) -> char* {
        off = (off + 255) & ~(size_t)255;
        char* p = ws + off;
        off += bytes;
        return p;
    };
    typedef unsigned short us;
    // activations
    us* xb   = (us*)alloc((size_t)M * DINP * 2);
    us* e0   = (us*)alloc((size_t)M * H * 2);
    us* e1   = (us*)alloc((size_t)M * H * 2);
    us* e2   = (us*)alloc((size_t)M * H * 2);
    us* zb   = (us*)alloc((size_t)7 * M * L * 2);
    us* hbuf = (us*)alloc((size_t)M * H * 2);
    us* tbuf = (us*)alloc((size_t)M * H * 2);
    // transposed bf16 weights: Wt[N_pad x K_pad]
    us* w0t   = (us*)alloc((size_t)H * DINP * 2);          // [512 x 800]
    us* encWt = (us*)alloc((size_t)3 * H * H * 2);         // [512 x 512] x3
    us* dWmuT = (us*)alloc((size_t)7 * LP * H * 2);        // [64 x 512] x7
    us* dWsgT = (us*)alloc((size_t)7 * LP * H * 2);
    us* tWhT  = (us*)alloc((size_t)6 * H * L * 2);         // [512 x 32] x6
    us* tWmuT = (us*)alloc((size_t)6 * LP * H * 2);        // [64 x 512] x6
    us* tWsgT = (us*)alloc((size_t)6 * LP * H * 2);
    us* rpW1T = (us*)alloc((size_t)3 * H * L * 2);         // [512 x 32] x3
    us* rpW2T = (us*)alloc((size_t)3 * H * H * 2);         // [512 x 512] x3
    us* rqW1T = (us*)alloc((size_t)3 * H * H * 2);
    us* rqW2T = (us*)alloc((size_t)3 * H * H * 2);
    us* dcW1T = (us*)alloc((size_t)4 * H * L * 2);         // [512 x 32] x4
    us* dcW2T = (us*)alloc((size_t)4 * DOUTP * H * 2);     // [832 x 512] x4
    // padded biases (f32)
    float* pb_dmu = (float*)alloc((size_t)7 * LP * 4);
    float* pb_dsg = (float*)alloc((size_t)7 * LP * 4);
    float* pb_tmu = (float*)alloc((size_t)6 * LP * 4);
    float* pb_tsg = (float*)alloc((size_t)6 * LP * 4);
    float* pb_db2 = (float*)alloc((size_t)4 * DOUTP * 4);
    // f32 scratch
    float* muq    = (float*)alloc((size_t)M * LP * 4);
    float* sqh    = (float*)alloc((size_t)M * LP * 4);
    float* mup    = (float*)alloc((size_t)M * LP * 4);
    float* spb    = (float*)alloc((size_t)M * LP * 4);
    float* pl     = (float*)alloc((size_t)M * 4);
    float* plq    = (float*)alloc((size_t)M * 4);
    float* probs  = (float*)alloc((size_t)7 * M * 4);
    float* klroot = (float*)alloc((size_t)M * 4);
    float* klnode = (float*)alloc((size_t)M * 4);
    float* kldec  = (float*)alloc((size_t)M * 4);
    float* rec    = (float*)alloc((size_t)M * 4);
    float* rbuf   = (float*)alloc((size_t)M * DOUTP * 4);

    // ---- init accumulators (harness does not re-zero between replays) ----
    zero3_k<<<M / 256, 256, 0, stream>>>(klnode, kldec, rec, M);

    // ---- convert x (K-padded) and all weights (transposed + padded) ----
    {
        long n = (long)M * DINP;
        cvt_pad_k<<<(unsigned)((n + 255) / 256), 256, 0, stream>>>(x, xb, M, DIN, M, DINP);
    }
    cvt_trans(enc_W0,    w0t,   1, DIN, H, DINP, H, stream);
    cvt_trans(enc_W,     encWt, 3, H, H, H, H, stream);
    cvt_trans(dense_Wmu, dWmuT, 7, H, L, H, LP, stream);
    cvt_trans(dense_Wsg, dWsgT, 7, H, L, H, LP, stream);
    cvt_trans(trans_Wh,  tWhT,  6, L, H, L, H, stream);
    cvt_trans(trans_Wmu, tWmuT, 6, H, L, H, LP, stream);
    cvt_trans(trans_Wsg, tWsgT, 6, H, L, H, LP, stream);
    cvt_trans(routp_W1,  rpW1T, 3, L, H, L, H, stream);
    cvt_trans(routp_W2,  rpW2T, 3, H, H, H, H, stream);
    cvt_trans(routq_W1,  rqW1T, 3, H, H, H, H, stream);
    cvt_trans(routq_W2,  rqW2T, 3, H, H, H, H, stream);
    cvt_trans(dec_W1,    dcW1T, 4, L, H, L, H, stream);
    cvt_trans(dec_W2,    dcW2T, 4, H, DIN, H, DOUTP, stream);
    bias_pad(dense_bmu, pb_dmu, 7, L, LP, stream);
    bias_pad(dense_bsg, pb_dsg, 7, L, LP, stream);
    bias_pad(trans_bmu, pb_tmu, 6, L, LP, stream);
    bias_pad(trans_bsg, pb_tsg, 6, L, LP, stream);
    bias_pad(dec_b2,    pb_db2, 4, DIN, DOUTP, stream);

    // ---- encoder chain (leaky-relu -> bf16) ----
    gemm(1, xb,   w0t,               enc_b0,        tbuf, M, H, DINP, stream);
    gemm(1, tbuf, encWt + 0 * H * H, enc_b + 0 * H, e0,   M, H, H,    stream);
    gemm(1, e0,   encWt + 1 * H * H, enc_b + 1 * H, e1,   M, H, H,    stream);
    gemm(1, e1,   encWt + 2 * H * H, enc_b + 2 * H, e2,   M, H, H,    stream);

    static const int depth_of[7] = {0, 1, 1, 2, 2, 2, 2};
    for (int k = 0; k < 7; k++) {
        int dl = depth_of[k];
        const us* d = (dl == 0) ? e2 : (dl == 1) ? e1 : e0;   // encoders[2 - dl]
        us* zk = zb + (size_t)k * M * L;

        gemm(0, d, dWmuT + (size_t)k * LP * H, pb_dmu + k * LP, muq, M, LP, H, stream);
        gemm(2, d, dWsgT + (size_t)k * LP * H, pb_dsg + k * LP, sqh, M, LP, H, stream);

        if (k > 0) {
            int j = k - 1;
            const us* zp = zb + (size_t)((k - 1) / 2) * M * L;
            gemm(1, zp,   tWhT  + (size_t)j * H * L,  trans_bh + j * H, hbuf, M, H,  L, stream);
            gemm(0, hbuf, tWmuT + (size_t)j * LP * H, pb_tmu + j * LP,  mup,  M, LP, H, stream);
            gemm(2, hbuf, tWsgT + (size_t)j * LP * H, pb_tsg + j * LP,  spb,  M, LP, H, stream);
        }

        node_stats_k<<<M / 256, 256, 0, stream>>>(
            muq, sqh,
            (k > 0) ? mup : nullptr, (k > 0) ? spb : nullptr,
            (k > 0) ? probs + (size_t)k * M : nullptr,
            (k == 0) ? klroot : klnode,
            zk, k, (k == 0) ? 1 : 0, M);

        if (k < 3) {
            // router p on z
            gemm(1, zk,   rpW1T + (size_t)k * H * L, routp_b1 + k * H, tbuf, M, H, L, stream);
            gemm(1, tbuf, rpW2T + (size_t)k * H * H, routp_b2 + k * H, hbuf, M, H, H, stream);
            rowdot_sigmoid_k<<<M / 8, 256, 0, stream>>>(hbuf, routp_W3 + k * H, routp_b3 + k, pl, M, H);
            // router q on d
            gemm(1, d,    rqW1T + (size_t)k * H * H, routq_b1 + k * H, tbuf, M, H, H, stream);
            gemm(1, tbuf, rqW2T + (size_t)k * H * H, routq_b2 + k * H, hbuf, M, H, H, stream);
            rowdot_sigmoid_k<<<M / 8, 256, 0, stream>>>(hbuf, routq_W3 + k * H, routq_b3 + k, plq, M, H);

            router_k<<<M / 256, 256, 0, stream>>>(
                pl, plq, (k > 0) ? probs + (size_t)k * M : nullptr,
                kldec, probs + (size_t)(2 * k + 1) * M, probs + (size_t)(2 * k + 2) * M, M);
        } else {
            int j = k - 3;
            gemm(1, zk,   dcW1T + (size_t)j * H * L,     dec_b1 + j * H,   hbuf, M, H,     L, stream);
            gemm(3, hbuf, dcW2T + (size_t)j * DOUTP * H, pb_db2 + j * DOUTP, rbuf, M, DOUTP, H, stream);
            bce_k<<<M / 8, 256, 0, stream>>>(x, rbuf, probs + (size_t)k * M, rec, M, DIN, DOUTP);
        }
    }

    final_k<<<M / 256, 256, 0, stream>>>(klnode, kldec, klroot, rec, (float*)d_out, M);
}
